// PartialTripletLoss_39350490366077
// MI455X (gfx1250) — compile-verified, matrix operand order checked by reading, not run on hardware
//
#include <hip/hip_runtime.h>
#include <hip/hip_bf16.h>
#include <math.h>

// ---------------------------------------------------------------------------
// PartialTripletLoss for MI455X (gfx1250, wave32, WMMA).
//
// Heavy op: gram[n,m,p] = sum_d x[n,d,p]*x[m,d,p]  -> 6x (2048x2048x1024) GEMM
// Strategy: repack fp32 -> bf16 (hi,lo) split, 3x v_wmma_f32_16x16x32_bf16 per
// 32-wide K chunk (hi*hi + hi*lo + lo*hi) for ~fp32 accuracy at bf16-WMMA rate.
// 48 MiB of operands live in the 192 MB L2, so the tile phase is L2-resident.
// Hardest-pos/neg row reductions are done with u32 atomics on float bit
// patterns (valid: all distances >= 0), avoiding a 16 MiB dist matrix.
// ---------------------------------------------------------------------------

typedef __attribute__((ext_vector_type(16))) __bf16 v16bf;
typedef __attribute__((ext_vector_type(8)))  __bf16 bf16x8;
typedef __attribute__((ext_vector_type(8)))  float  v8f;

#define MARGIN_F 0.3f

// ---------------------------------------------------------------------------
// Kernel 0: init per-row hardest-positive (max, start at 0.0) and
//           hardest-negative (min, start at +inf) bit arrays.
// ---------------------------------------------------------------------------
__global__ void PTL_init(unsigned* __restrict__ ap_bits,
                         unsigned* __restrict__ an_bits, int n) {
  int i = blockIdx.x * blockDim.x + threadIdx.x;
  if (i < n) {
    ap_bits[i] = 0u;           // 0.0f  (dist >= 0 always; diagonal included)
    an_bits[i] = 0x7F800000u;  // +inf
  }
}

// ---------------------------------------------------------------------------
// Kernel 1: repack inputs[n,d,P] (fp32, P-strided) into per-part row-major
// bf16 hi/lo matrices [P][n][d], and compute sqn[p][i] = sum_d x^2.
// One wave32 per (i,p); strided reads once, contiguous bf16 writes.
// ---------------------------------------------------------------------------
__global__ __launch_bounds__(256) void PTL_repack(
    const float* __restrict__ x, __bf16* __restrict__ hi,
    __bf16* __restrict__ lo, float* __restrict__ sqn,
    int n, int d, int P) {
  int wave = (int)((blockIdx.x * blockDim.x + threadIdx.x) >> 5);
  int lane = threadIdx.x & 31;
  if (wave >= n * P) return;
  int p = wave % P;
  int i = wave / P;

  const float* src = x + (size_t)i * d * P + p;
  __bf16* hdst = hi + ((size_t)p * n + i) * d;
  __bf16* ldst = lo + ((size_t)p * n + i) * d;

  float s = 0.f;
  for (int dd = lane; dd < d; dd += 32) {
    float v = src[(size_t)dd * P];
    __bf16 h = (__bf16)v;                 // RNE truncate to bf16
    __bf16 l = (__bf16)(v - (float)h);    // residual (second bf16 "digit")
    hdst[dd] = h;
    ldst[dd] = l;
    s += v * v;
  }
#pragma unroll
  for (int off = 16; off > 0; off >>= 1) s += __shfl_xor(s, off, 32);
  if (lane == 0) sqn[(size_t)p * n + i] = s;
}

// ---------------------------------------------------------------------------
// Kernel 2: one wave32 per 16x16 output tile. For each part p, accumulate the
// Gram tile with 3 bf16 WMMAs per 32-wide K chunk, then fold the masked
// per-part sqrt-distance into per-lane accumulators. Finally atomically
// reduce into per-row hardest-pos/neg arrays.
//
// ISA layouts used (CDNA5 ISA):
//   A (16x32 bf16): lane L (m = L&15, half = L>>4) holds
//       e<8  -> K = 8*half + e          (contiguous 16B chunk)
//       e>=8 -> K = 8*half + 16 + (e-8) (contiguous 16B chunk)
//   B (32x16 bf16): assumed symmetric layout with N in the lane role,
//       so a B tile load == an A tile load from the j-tile rows.
//   C/D f32: element r of v8f in lane L -> row r + 8*half, col L&15.
// ---------------------------------------------------------------------------
__global__ __launch_bounds__(32) void PTL_tile(
    const __bf16* __restrict__ Ahi, const __bf16* __restrict__ Alo,
    const float* __restrict__ sqn, const int* __restrict__ targets,
    const int* __restrict__ plab,
    unsigned* __restrict__ ap_bits, unsigned* __restrict__ an_bits,
    int n, int d, int P, int ntj) {
  int tile = blockIdx.x;
  int ti = tile / ntj;
  int tj = tile % ntj;
  int I0 = ti * 16, J0 = tj * 16;

  int lane = threadIdx.x & 31;
  int half = lane >> 4;
  int lrow = lane & 15;

  // This lane's C column and the 8 C rows it owns.
  int jcol = J0 + lrow;
  int jS = plab[2 * jcol], jE = plab[2 * jcol + 1];
  int tJ = targets[jcol];
  int iS[8], iE[8], tI[8];
#pragma unroll
  for (int r = 0; r < 8; ++r) {
    int i = I0 + r + 8 * half;
    iS[r] = plab[2 * i];
    iE[r] = plab[2 * i + 1];
    tI[r] = targets[i];
  }

  float dacc[8], nacc[8];
#pragma unroll
  for (int r = 0; r < 8; ++r) { dacc[r] = 0.f; nacc[r] = 0.f; }

  for (int p = 0; p < P; ++p) {
    const __bf16* base_h = Ahi + ((size_t)p * n) * d;
    const __bf16* base_l = Alo + ((size_t)p * n) * d;
    const __bf16* arow_h = base_h + (size_t)(I0 + lrow) * d;
    const __bf16* arow_l = base_l + (size_t)(I0 + lrow) * d;
    const __bf16* brow_h = base_h + (size_t)(J0 + lrow) * d;
    const __bf16* brow_l = base_l + (size_t)(J0 + lrow) * d;
    __builtin_prefetch(brow_h, 0, 3);  // global_prefetch of B rows

    v8f acc = {};
    for (int k0 = 0; k0 < d; k0 += 32) {
      int base = k0 + 8 * half;
      bf16x8 a0 = *(const bf16x8*)(arow_h + base);
      bf16x8 a1 = *(const bf16x8*)(arow_h + base + 16);
      bf16x8 a0l = *(const bf16x8*)(arow_l + base);
      bf16x8 a1l = *(const bf16x8*)(arow_l + base + 16);
      bf16x8 b0 = *(const bf16x8*)(brow_h + base);
      bf16x8 b1 = *(const bf16x8*)(brow_h + base + 16);
      bf16x8 b0l = *(const bf16x8*)(brow_l + base);
      bf16x8 b1l = *(const bf16x8*)(brow_l + base + 16);

      v16bf ah, al, bh, bl;
#pragma unroll
      for (int e = 0; e < 8; ++e) {
        ah[e] = a0[e];  ah[8 + e] = a1[e];
        al[e] = a0l[e]; al[8 + e] = a1l[e];
        bh[e] = b0[e];  bh[8 + e] = b1[e];
        bl[e] = b0l[e]; bl[8 + e] = b1l[e];
      }
      // fp32-accurate split product: hi*hi + hi*lo + lo*hi (drop lo*lo)
      acc = __builtin_amdgcn_wmma_f32_16x16x32_bf16(false, ah, false, bh,
                                                    (short)0, acc, false, false);
      acc = __builtin_amdgcn_wmma_f32_16x16x32_bf16(false, ah, false, bl,
                                                    (short)0, acc, false, false);
      acc = __builtin_amdgcn_wmma_f32_16x16x32_bf16(false, al, false, bh,
                                                    (short)0, acc, false, false);
    }

    bool fj = (jS <= p) && (jE >= p);
    float sqj = sqn[(size_t)p * n + jcol];
#pragma unroll
    for (int r = 0; r < 8; ++r) {
      bool fi = (iS[r] <= p) && (iE[r] >= p);
      if (fi && fj) {
        int i = I0 + r + 8 * half;
        float sqi = sqn[(size_t)p * n + i];
        float g = acc[r];
        float sq = fmaxf(sqi + sqj - 2.0f * g, 1e-12f);
        dacc[r] += sqrtf(sq);
        nacc[r] += 1.0f;
      }
    }
  }

  // Fold into per-row hardest positive / negative (num > 0 guaranteed).
#pragma unroll
  for (int r = 0; r < 8; ++r) {
    float dist = dacc[r] / nacc[r];
    unsigned bits = __float_as_uint(dist);  // dist >= 0 -> uint order == float order
    int i = I0 + r + 8 * half;
    if (tI[r] == tJ) atomicMax(&ap_bits[i], bits);
    else             atomicMin(&an_bits[i], bits);
  }
}

// ---------------------------------------------------------------------------
// Kernel 3: final scalar reduction over n rows.
// out = [loss, prec, mean(dist_ap), mean(dist_an)]
// ---------------------------------------------------------------------------
__global__ __launch_bounds__(256) void PTL_final(
    const unsigned* __restrict__ ap_bits, const unsigned* __restrict__ an_bits,
    float* __restrict__ out, int n) {
  __shared__ float sl[256], sp[256], sa[256], sb[256];
  int t = threadIdx.x;
  float ls = 0.f, ps = 0.f, as_ = 0.f, bs = 0.f;
  for (int i = t; i < n; i += 256) {
    float ap = __uint_as_float(ap_bits[i]);
    float an = __uint_as_float(an_bits[i]);
    ls += fmaxf(ap - an + MARGIN_F, 0.f);
    ps += (an > ap) ? 1.f : 0.f;
    as_ += ap;
    bs += an;
  }
  sl[t] = ls; sp[t] = ps; sa[t] = as_; sb[t] = bs;
  __syncthreads();
  for (int s = 128; s > 0; s >>= 1) {
    if (t < s) {
      sl[t] += sl[t + s]; sp[t] += sp[t + s];
      sa[t] += sa[t + s]; sb[t] += sb[t + s];
    }
    __syncthreads();
  }
  if (t == 0) {
    float inv = 1.0f / (float)n;
    out[0] = sl[0] * inv;
    out[1] = sp[0] * inv;
    out[2] = sa[0] * inv;
    out[3] = sb[0] * inv;
  }
}

// ---------------------------------------------------------------------------
extern "C" void kernel_launch(void* const* d_in, const int* in_sizes, int n_in,
                              void* d_out, int out_size, void* d_ws,
                              size_t ws_size, hipStream_t stream) {
  (void)n_in; (void)out_size; (void)ws_size;
  const float* inputs  = (const float*)d_in[0];  // [n, d, P] fp32
  const int*   targets = (const int*)d_in[1];    // [n] int32
  const int*   plab    = (const int*)d_in[2];    // [n, 2] int32

  const int n = in_sizes[1];
  const int P = 6;                       // fixed by the reference
  const int d = in_sizes[0] / (n * P);   // 1024

  // Workspace layout: bf16 hi | bf16 lo | sqn f32 [P][n] | ap u32 [n] | an u32 [n]
  size_t szbf = (size_t)P * n * d * sizeof(__bf16);  // 24 MiB each
  char* base = (char*)d_ws;
  __bf16*   hi  = (__bf16*)base;
  __bf16*   lo  = (__bf16*)(base + szbf);
  float*    sqn = (float*)(base + 2 * szbf);
  unsigned* ap  = (unsigned*)(base + 2 * szbf + (size_t)P * n * sizeof(float));
  unsigned* an  = ap + n;

  PTL_init<<<(n + 255) / 256, 256, 0, stream>>>(ap, an, n);

  int waves = n * P;                     // one wave32 per (i, p)
  PTL_repack<<<(waves * 32 + 255) / 256, 256, 0, stream>>>(
      inputs, hi, lo, sqn, n, d, P);

  int nt = n / 16;                       // 128 tiles per dimension
  PTL_tile<<<nt * nt, 32, 0, stream>>>(hi, lo, sqn, targets, plab, ap, an,
                                       n, d, P, nt);

  PTL_final<<<1, 256, 0, stream>>>(ap, an, (float*)d_out, n);
}
